// DiffusionCNN_27315992003059
// MI455X (gfx1250) — matrix operand order repaired and checked by reading, not compiled, not run on hardware
//
#include <hip/hip_runtime.h>
#include <hip/hip_bf16.h>
#include <math.h>

// ---------------------------------------------------------------------------
// DiffusionCNN on MI455X (gfx1250): sparse 3^3 conv layers lowered to
// v_wmma_f32_16x16x32_f16. f16 activations/weights, f32 accumulate.
// Round 3: layer-2 gather staged with global_load_async_to_lds_b128
// (ASYNCcnt-tracked DMA, no VGPR round-trip), double-buffered LDS.
// ---------------------------------------------------------------------------

typedef _Float16 v8h  __attribute__((ext_vector_type(8)));
typedef _Float16 v16h __attribute__((ext_vector_type(16)));
typedef float    v8f  __attribute__((ext_vector_type(8)));

union ABReg { v16h v; v8h h[2]; };

#define KOFF 27
#define KPAD 28
#define CHID 128

__device__ __forceinline__ float silu_f(float x) {
    return x * (1.0f / (1.0f + __expf(-x)));
}

__device__ __forceinline__ v8f wmma16(v16h a, v16h b, v8f c) {
    // D = A(16x32 f16) * B(32x16 f16) + C(16x16 f32)
    return __builtin_amdgcn_wmma_f32_16x16x32_f16(
        /*neg_a=*/false, a, /*neg_b=*/false, b,
        /*c_mod=*/(short)0, c, /*reuse_a=*/false, /*reuse_b=*/false);
}

// Async DMA: copy 32 contiguous bytes global -> LDS (2 x B128, ASYNCcnt).
// INST_OFFSET adds to both the LDS and the global address (ISA 15.18.3).
__device__ __forceinline__ void async_copy32(unsigned lds_off, const void* gptr) {
    asm volatile("global_load_async_to_lds_b128 %0, %1, off"
                 :: "v"(lds_off), "v"(gptr) : "memory");
    asm volatile("global_load_async_to_lds_b128 %0, %1, off offset:16"
                 :: "v"(lds_off), "v"(gptr) : "memory");
}

__device__ __forceinline__ void wait_async0() {
    asm volatile("s_wait_asynccnt 0" ::: "memory");
}

// D-tile (16x16 f32): elem e of lane L -> (row = (L<16?e:8+e), col = L%16).
__device__ __forceinline__ void store_silu_f16(_Float16* __restrict__ out,
                                               v8f acc, const float* __restrict__ bias,
                                               int m0, int ntile, int lane, int N) {
    const int col = ntile * 16 + (lane & 15);
    const float bv = bias[col];
    const int rbase = (lane < 16) ? 0 : 8;
#pragma unroll
    for (int e = 0; e < 8; ++e) {
        int m = m0 + rbase + e;
        if (m < N) out[(size_t)m * CHID + col] = (_Float16)silu_f(acc[e] + bv);
    }
}

// -------------------- prep: x = [feat, sincos(t)] padded to 8 ch ------------
__global__ void k_embed(const float* __restrict__ feat, const int* __restrict__ t,
                        _Float16* __restrict__ xpad8, int N) {
    int i = blockIdx.x * blockDim.x + threadIdx.x;
    if (i > N) return;
    _Float16* row = xpad8 + (size_t)i * 8;
    if (i == N) {                       // sentinel row -> zeros
#pragma unroll
        for (int c = 0; c < 8; ++c) row[c] = (_Float16)0.0f;
        return;
    }
    float tv = (float)t[i];
    float a0 = tv * 3.14159265358979323846f;
    float a1 = a0 * 2.0f, a2 = a0 * 4.0f;
    row[0] = (_Float16)feat[i];
    row[1] = (_Float16)sinf(a0);
    row[2] = (_Float16)sinf(a1);
    row[3] = (_Float16)sinf(a2);
    row[4] = (_Float16)cosf(a0);
    row[5] = (_Float16)cosf(a1);
    row[6] = (_Float16)cosf(a2);
    row[7] = (_Float16)0.0f;
}

// -------------------- prep: padded rulebook (28 x Npad, sentinel N) ---------
__global__ void k_prep_nidx(const int* __restrict__ nidx, int* __restrict__ nidxP,
                            int N, int Npad) {
    int m = blockIdx.x * blockDim.x + threadIdx.x;
    int k = blockIdx.y;
    if (m >= Npad) return;
    int v = N;
    if (k < KOFF && m < N) v = nidx[k * N + m];
    nidxP[(size_t)k * Npad + m] = v;
}

// -------------------- prep: weights -> WMMA-B-ready f16 layouts -------------
// B 32x16 f16 tile layout (ISA 7.12.2): lane holds col = lane%16,
// K = (e>=8 ? 16 : 0) + (lane>=16 ? 8 : 0) + (e&7), e = 0..15; each lane's 16 halves contiguous.

// W1 packed: K = 4 offsets x 8 in-ch (27 offsets padded to 28).
__global__ void k_prep_w1(const float* __restrict__ W1, _Float16* __restrict__ w1b) {
    int i = blockIdx.x * blockDim.x + threadIdx.x;
    if (i >= 7 * 8 * 512) return;
    int e = i & 15, lane = (i >> 4) & 31, nt = (i >> 9) & 7, g = i >> 12;
    int K  = (e & 8) * 2 + ((lane >= 16) ? 8 : 0) + (e & 7);
    int og = K >> 3, c = K & 7;                 // offset-in-group, in-channel
    int kb = g * 4 + og;
    int col = nt * 16 + (lane & 15);
    float v = 0.0f;
    if (kb < KOFF && c < 7) v = W1[((size_t)kb * 7 + c) * CHID + col];
    w1b[i] = (_Float16)v;
}

// w2b[k][kc][nt][lane][e]
__global__ void k_prep_w2(const float* __restrict__ W2, _Float16* __restrict__ w2b) {
    int i = blockIdx.x * blockDim.x + threadIdx.x;
    if (i >= KOFF * 4 * 8 * 512) return;
    int e = i & 15, lane = (i >> 4) & 31, nt = (i >> 9) & 7;
    int kc = (i >> 12) & 3, k = i >> 14;
    int kk = (e & 8) * 2 + ((lane >= 16) ? 8 : 0) + (e & 7);
    int ci = kc * 32 + kk;
    int col = nt * 16 + (lane & 15);
    w2b[i] = (_Float16)W2[((size_t)k * CHID + ci) * CHID + col];
}

// w3b[kc][nt][lane][e]
__global__ void k_prep_w3(const float* __restrict__ W3, _Float16* __restrict__ w3b) {
    int i = blockIdx.x * blockDim.x + threadIdx.x;
    if (i >= 4 * 8 * 512) return;
    int e = i & 15, lane = (i >> 4) & 31, nt = (i >> 9) & 7, kc = i >> 12;
    int kk = (e & 8) * 2 + ((lane >= 16) ? 8 : 0) + (e & 7);
    int ci = kc * 32 + kk;
    int col = nt * 16 + (lane & 15);
    w3b[i] = (_Float16)W3[(size_t)ci * CHID + col];
}

__global__ void k_zero_h1row(_Float16* __restrict__ h1, int N) {
    int i = threadIdx.x;
    if (i < CHID) h1[(size_t)N * CHID + i] = (_Float16)0.0f;
}

// -------------------- layer 1: sparse conv, Cin=8(padded), WMMA -------------
// A-tile (16x32 f16): lane<16 row=lane, K{0-7,16-23}; lane>=16 row=lane-16,
// K{8-15,24-31}. K = og*8 + c -> lane<16 gathers offsets {4g, 4g+2},
// lane>=16 gathers {4g+1, 4g+3}. Unconditional loads via padded rulebook.
__global__ __launch_bounds__(128)
void k_layer1(const _Float16* __restrict__ xpad8, const _Float16* __restrict__ w1b,
              const float* __restrict__ b1, const int* __restrict__ nidxP,
              _Float16* __restrict__ h1, int N, int Npad) {
    const int tid = threadIdx.x;
    const int wave = tid >> 5, lane = tid & 31;
    const int m0 = blockIdx.x * 16;
    const bool hi = lane >= 16;
    const int mrow = m0 + (lane & 15);
    const int nt0 = wave * 2;
    v8f acc0 = {}, acc1 = {};
#pragma unroll
    for (int g = 0; g < 7; ++g) {
        int k0 = g * 4 + (hi ? 1 : 0);
        int k1 = g * 4 + (hi ? 3 : 2);
        int idx0 = nidxP[(size_t)k0 * Npad + mrow];
        int idx1 = nidxP[(size_t)k1 * Npad + mrow];
        ABReg a;
        a.h[0] = *(const v8h*)(xpad8 + (size_t)idx0 * 8);
        a.h[1] = *(const v8h*)(xpad8 + (size_t)idx1 * 8);
        v16h bv0 = *(const v16h*)(w1b + (((size_t)g * 8 + nt0)     * 32 + lane) * 16);
        v16h bv1 = *(const v16h*)(w1b + (((size_t)g * 8 + nt0 + 1) * 32 + lane) * 16);
        acc0 = wmma16(a.v, bv0, acc0);
        acc1 = wmma16(a.v, bv1, acc1);
    }
    store_silu_f16(h1, acc0, b1, m0, nt0,     lane, N);
    store_silu_f16(h1, acc1, b1, m0, nt0 + 1, lane, N);
}

// -------------------- layer 2: sparse conv 128->128 (dominant) --------------
// Block = 32 output rows; wave = 2 M-tiles x 2 N-tiles = 4 accumulators.
// Double-buffered LDS; the k+1 gather is issued as async-to-LDS DMA
// (global_load_async_to_lds_b128) before the 16 WMMAs of offset k, then
// s_wait_asynccnt 0 + barrier. LDS row stride 136 halves (272B) keeps the
// 16-lane ds_load_b128 A-reads bank-conflict free.
__global__ __launch_bounds__(128)
void k_layer2(const _Float16* __restrict__ h1, const _Float16* __restrict__ w2b,
              const float* __restrict__ b2, const int* __restrict__ nidxP,
              _Float16* __restrict__ h2, int N, int Npad) {
    __shared__ _Float16 ldsA[2][32 * 136];
    const int tid = threadIdx.x;
    const int wave = tid >> 5, lane = tid & 31;
    const int m0 = blockIdx.x * 32;
    const bool hi = lane >= 16;
    const int nt0 = wave * 2;
    // staging: 128 threads cover 32 rows x 8 chunks of 32B (2 rows per thread)
    const int r0 = tid >> 3;            // 0..15
    const int r1 = r0 + 16;             // 16..31
    const int ch = (tid & 7) * 16;      // halves
    const unsigned ldsBase0 = (unsigned)(size_t)&ldsA[0][0];
    const unsigned ldsBase1 = (unsigned)(size_t)&ldsA[1][0];
    const unsigned la = (unsigned)((r0 * 136 + ch) * 2);   // byte offsets in buffer
    const unsigned lb = (unsigned)((r1 * 136 + ch) * 2);
    v8f acc[4] = {{}, {}, {}, {}};      // [mt][nt]

    {   // prologue: async gather for k = 0 into buffer 0
        int idxA = nidxP[m0 + r0];
        int idxB = nidxP[m0 + r1];
        async_copy32(ldsBase0 + la, h1 + (size_t)idxA * CHID + ch);
        async_copy32(ldsBase0 + lb, h1 + (size_t)idxB * CHID + ch);
    }
    wait_async0();
    __syncthreads();

    for (int k = 0; k < KOFF; ++k) {
        const int cur = k & 1;
        if (k + 1 < KOFF) {             // async gather of next offset -> back buffer
            const unsigned dst = cur ? ldsBase0 : ldsBase1;
            int idxA = nidxP[(size_t)(k + 1) * Npad + m0 + r0];
            int idxB = nidxP[(size_t)(k + 1) * Npad + m0 + r1];
            async_copy32(dst + la, h1 + (size_t)idxA * CHID + ch);
            async_copy32(dst + lb, h1 + (size_t)idxB * CHID + ch);
        }
#pragma unroll
        for (int kc = 0; kc < 4; ++kc) {
            v16h bv0 = *(const v16h*)(w2b + ((((size_t)k * 4 + kc) * 8 + nt0)     * 32 + lane) * 16);
            v16h bv1 = *(const v16h*)(w2b + ((((size_t)k * 4 + kc) * 8 + nt0 + 1) * 32 + lane) * 16);
#pragma unroll
            for (int mt = 0; mt < 2; ++mt) {
                ABReg a;
                const _Float16* ap =
                    &ldsA[cur][(mt * 16 + (lane & 15)) * 136 + kc * 32 + (hi ? 8 : 0)];
                a.h[0] = *(const v8h*)(ap);
                a.h[1] = *(const v8h*)(ap + 16);
                acc[mt * 2 + 0] = wmma16(a.v, bv0, acc[mt * 2 + 0]);
                acc[mt * 2 + 1] = wmma16(a.v, bv1, acc[mt * 2 + 1]);
            }
        }
        if (k + 1 < KOFF) wait_async0();   // own DMAs landed in LDS
        __syncthreads();                    // visible to all waves
    }
    store_silu_f16(h2, acc[0], b2, m0,      nt0,     lane, N);
    store_silu_f16(h2, acc[1], b2, m0,      nt0 + 1, lane, N);
    store_silu_f16(h2, acc[2], b2, m0 + 16, nt0,     lane, N);
    store_silu_f16(h2, acc[3], b2, m0 + 16, nt0 + 1, lane, N);
}

// -------------------- layers 3+4 fused: 1x1 conv + 128->1 projection --------
__global__ __launch_bounds__(128)
void k_layer34(const _Float16* __restrict__ h2, const _Float16* __restrict__ w3b,
               const float* __restrict__ b3, const float* __restrict__ W4,
               const float* __restrict__ b4, float* __restrict__ out, int N) {
    __shared__ float red[16];
    const int tid = threadIdx.x;
    const int wave = tid >> 5, lane = tid & 31;
    const int m0 = blockIdx.x * 16;
    const bool hi = lane >= 16;
    const int nt0 = wave * 2;
    int rowA = m0 + (lane & 15);
    if (rowA >= N) rowA = N - 1;      // clamp; stores guarded below
    v8f acc0 = {}, acc1 = {};
#pragma unroll
    for (int kc = 0; kc < 4; ++kc) {
        ABReg a;
        const _Float16* ap = h2 + (size_t)rowA * CHID + kc * 32 + (hi ? 8 : 0);
        a.h[0] = *(const v8h*)(ap);
        a.h[1] = *(const v8h*)(ap + 16);
        v16h bv0 = *(const v16h*)(w3b + (((size_t)kc * 8 + nt0)     * 32 + lane) * 16);
        v16h bv1 = *(const v16h*)(w3b + (((size_t)kc * 8 + nt0 + 1) * 32 + lane) * 16);
        acc0 = wmma16(a.v, bv0, acc0);
        acc1 = wmma16(a.v, bv1, acc1);
    }
    // layer 4: out[m] = sum_c silu(h3[m][c]) * W4[c] + b4
    const int colL = lane & 15;
    float b3c0 = b3[nt0 * 16 + colL], b3c1 = b3[(nt0 + 1) * 16 + colL];
    float w40  = W4[nt0 * 16 + colL], w41  = W4[(nt0 + 1) * 16 + colL];
    float part[8];
#pragma unroll
    for (int e = 0; e < 8; ++e) {
        float p0 = silu_f(acc0[e] + b3c0);
        float p1 = silu_f(acc1[e] + b3c1);
        part[e] = p0 * w40 + p1 * w41;
    }
#pragma unroll
    for (int e = 0; e < 8; ++e) {
#pragma unroll
        for (int m = 1; m < 16; m <<= 1) part[e] += __shfl_xor(part[e], m, 32);
    }
    if (tid < 16) red[tid] = 0.0f;
    __syncthreads();
    if (lane == 0) {
#pragma unroll
        for (int e = 0; e < 8; ++e) atomicAdd(&red[e], part[e]);
    } else if (lane == 16) {
#pragma unroll
        for (int e = 0; e < 8; ++e) atomicAdd(&red[8 + e], part[e]);
    }
    __syncthreads();
    if (tid < 16) {
        int m = m0 + tid;
        if (m < N) out[m] = red[tid] + b4[0];
    }
}

// ---------------------------------------------------------------------------
extern "C" void kernel_launch(void* const* d_in, const int* in_sizes, int n_in,
                              void* d_out, int out_size, void* d_ws, size_t ws_size,
                              hipStream_t stream) {
    const float* feat = (const float*)d_in[0];
    const int*   tin  = (const int*)d_in[1];
    const int*   nidx = (const int*)d_in[2];
    const float* W1   = (const float*)d_in[3];
    const float* b1   = (const float*)d_in[4];
    const float* W2   = (const float*)d_in[5];
    const float* b2   = (const float*)d_in[6];
    const float* W3   = (const float*)d_in[7];
    const float* b3   = (const float*)d_in[8];
    const float* W4   = (const float*)d_in[9];
    const float* b4   = (const float*)d_in[10];
    float* out = (float*)d_out;
    const int N = in_sizes[0];
    const int Npad = ((N + 31) / 32) * 32;   // covers 32-row layer-2 tiles
    (void)n_in; (void)out_size; (void)ws_size;

    char* ws = (char*)d_ws;
    size_t off = 0;
    auto take = [&](size_t bytes) {
        char* p = ws + off;
        off += (bytes + 255) & ~(size_t)255;
        return p;
    };
    _Float16* xpad8 = (_Float16*)take((size_t)(N + 1) * 8 * sizeof(_Float16));
    _Float16* h1    = (_Float16*)take((size_t)(N + 1) * CHID * sizeof(_Float16));
    _Float16* h2    = (_Float16*)take((size_t)N * CHID * sizeof(_Float16));
    _Float16* w1b   = (_Float16*)take((size_t)7 * 8 * 512 * sizeof(_Float16));
    _Float16* w2b   = (_Float16*)take((size_t)KOFF * 4 * 8 * 512 * sizeof(_Float16));
    _Float16* w3b   = (_Float16*)take((size_t)4 * 8 * 512 * sizeof(_Float16));
    int*      nidxP = (int*)take((size_t)KPAD * Npad * sizeof(int));

    k_embed    <<<dim3((N + 1 + 255) / 256), dim3(256), 0, stream>>>(feat, tin, xpad8, N);
    k_prep_nidx<<<dim3((Npad + 255) / 256, KPAD), dim3(256), 0, stream>>>(nidx, nidxP, N, Npad);
    k_prep_w1  <<<dim3((7 * 8 * 512 + 255) / 256), dim3(256), 0, stream>>>(W1, w1b);
    k_prep_w2  <<<dim3((KOFF * 4 * 8 * 512 + 255) / 256), dim3(256), 0, stream>>>(W2, w2b);
    k_prep_w3  <<<dim3((4 * 8 * 512 + 255) / 256), dim3(256), 0, stream>>>(W3, w3b);
    k_zero_h1row<<<dim3(1), dim3(128), 0, stream>>>(h1, N);
    k_layer1   <<<dim3(Npad / 16), dim3(128), 0, stream>>>(xpad8, w1b, b1, nidxP, h1, N, Npad);
    k_layer2   <<<dim3(Npad / 32), dim3(128), 0, stream>>>(h1, w2b, b2, nidxP, h2, N, Npad);
    k_layer34  <<<dim3((N + 15) / 16), dim3(128), 0, stream>>>(h2, w3b, b3, W4, b4, out, N);
}
